// Corr_decoder_14869176779357
// MI455X (gfx1250) — matrix operand order repaired and checked by reading, not compiled, not run on hardware
//
#include <hip/hip_runtime.h>

#define B_   8
#define C_   256
#define H_   96
#define W_   128
#define MD_  4
#define ND_  9            // 2*MD+1 displacements per axis
#define NDG_ 3            // dy rows per workgroup (9 = 3 groups of 3)
#define TW_  16           // w-tile (WMMA M)
#define BP_  24           // x2 positions per row tile = TW_ + 2*MD
#define LDSTR 264         // padded channel stride in ushorts (breaks bank conflicts)
#define NT_  96           // threads per block = 3 waves, one dy per wave

typedef __bf16 v16bf __attribute__((ext_vector_type(16)));
typedef float  v8f   __attribute__((ext_vector_type(8)));
typedef unsigned int v4u __attribute__((ext_vector_type(4)));

union Frag {
    v16bf v;
    v4u   q[2];
};

__device__ __forceinline__ unsigned short f2bf(float f) {
    union { float f; unsigned u; } x;
    x.f = f;
    unsigned r = x.u + 0x7FFFu + ((x.u >> 16) & 1u);   // round-to-nearest-even
    return (unsigned short)(r >> 16);
}

// B-fragment load: lane holds column N = lane&15, K = kstep*32 + (hi?8:0) + {0..7,16..23}
// -> two ds_load_b128.
__device__ __forceinline__ void ldB(Frag& f, const unsigned short* p, int k) {
    f.q[0] = *(const v4u*)(p + k * 32);
    f.q[1] = *(const v4u*)(p + k * 32 + 16);
}

__launch_bounds__(NT_)
__global__ void corr_wmma_kernel(const float* __restrict__ x1,
                                 const float* __restrict__ x2,
                                 float* __restrict__ out) {
    // ~47 KB total -> ~6 blocks/WGP, ~4.5 waves/SIMD for latency hiding.
    __shared__ __align__(16) unsigned short sA[TW_ * LDSTR];           // 8.4 KB
    __shared__ __align__(16) unsigned short sB[NDG_ * BP_ * LDSTR];    // 37.1 KB
    __shared__ float sBand[NDG_][ND_ * TW_];                           // 1.7 KB

    const int tid    = threadIdx.x;
    const int lane   = tid & 31;
    const int wv     = tid >> 5;              // 0..2 -> local dy
    const int w0     = blockIdx.x * TW_;
    const int h      = blockIdx.y;
    const int b      = blockIdx.z / NDG_;
    const int dyBase = (blockIdx.z % NDG_) * NDG_;

    // ---- stage A: x1[b, :, h, w0:w0+16] -> bf16 LDS (pixel-major rows) ----
    {
        const float* src = x1 + ((size_t)(b * C_) * H_ + h) * W_ + w0;
        for (int idx = tid; idx < TW_ * C_; idx += NT_) {
            int m = idx & (TW_ - 1);
            int c = idx >> 4;
            sA[m * LDSTR + c] = f2bf(src[(size_t)c * (H_ * W_) + m]);
        }
    }
    // ---- stage B: x2 rows h+dyBase-4 .. +2, positions w0-4..w0+19, zero-pad ----
    {
        for (int idx = tid; idx < NDG_ * BP_ * C_; idx += NT_) {
            int p   = idx % BP_;
            int t   = idx / BP_;
            int c   = t % C_;
            int dyl = t / C_;
            int hy  = h + dyBase + dyl - MD_;
            int wx  = w0 - MD_ + p;
            float v = 0.0f;
            if ((unsigned)hy < (unsigned)H_ && (unsigned)wx < (unsigned)W_)
                v = x2[(((size_t)b * C_ + c) * H_ + hy) * W_ + wx];
            sB[(dyl * BP_ + p) * LDSTR + c] = f2bf(v);
        }
    }
    __syncthreads();

    const int hi = (lane >> 4) & 1;      // lane group (0: lanes 0-15, 1: lanes 16-31)
    const int nm = lane & 15;            // A: row M ; B: column N

    // ---- build all 8 A-fragments (shared across both chains) ----
    // ISA 16-bit A 16x32 layout: lanes 0-15 hold K = {0..7,16..23}, lanes 16-31
    // hold K = {8..15,24..31} for row M = lane&15.
    Frag a[8];
    {
        const unsigned short* ap = sA + nm * LDSTR + hi * 8;
        #pragma unroll
        for (int k = 0; k < 8; ++k) {
            a[k].q[0] = *(const v4u*)(ap + k * 32);
            a[k].q[1] = *(const v4u*)(ap + k * 32 + 16);
        }
    }

    // ---- one dy per wave: two independent 16x16x256 chains (n-tiles 0,1) ----
    const unsigned short* p0 = sB + ((size_t)(wv * BP_ + nm)) * LDSTR + hi * 8;
    const unsigned short* p1 = p0 + 16 * LDSTR;

    v8f acc0 = {};
    v8f acc1 = {};
    Frag f0[2], f1[2];
    ldB(f0[0], p0, 0);
    ldB(f1[0], p1, 0);

    // Software-pipelined K loop, double-buffered with constant indices (no movs).
    #pragma unroll
    for (int k = 0; k < 8; ++k) {
        const int cur = k & 1;
        const int nxt = cur ^ 1;
        if (k < 7) {
            ldB(f0[nxt], p0, k + 1);
            ldB(f1[nxt], p1, k + 1);
        }
        acc0 = __builtin_amdgcn_wmma_f32_16x16x32_bf16(
            false, a[k].v, false, f0[cur].v, (short)0, acc0, false, false);
        acc1 = __builtin_amdgcn_wmma_f32_16x16x32_bf16(
            false, a[k].v, false, f1[cur].v, (short)0, acc1, false, false);
    }

    // ---- band extraction into per-wave LDS staging ----
    // acc layout: lane 0-15 -> N=lane, VGPR r -> M=r ; lanes 16-31 -> M=r+8.
    // x2 position = w0 - 4 + col; dxi = col - m.
    float* band = sBand[wv];
    #pragma unroll
    for (int r = 0; r < 8; ++r) {
        int m    = r + hi * 8;
        int dxs0 = nm - m;           // n-tile 0: col = nm
        if (dxs0 >= 0 && dxs0 <= 2 * MD_)
            band[dxs0 * TW_ + m] = acc0[r];
        int dxs1 = nm + 16 - m;      // n-tile 1: col = nm + 16
        if (dxs1 >= 0 && dxs1 <= 2 * MD_)
            band[dxs1 * TW_ + m] = acc1[r];
    }
    __asm volatile("s_wait_dscnt 0" ::: "memory");

    // ---- scale + leaky ReLU + coalesced store (16 floats per (d,h) row) ----
    const float scale = 1.0f / (float)C_;
    const int dy = dyBase + wv;
    for (int i = lane; i < ND_ * TW_; i += 32) {
        int dxi = i >> 4;
        int m   = i & 15;
        float v = band[i] * scale;
        v = (v >= 0.0f) ? v : v * 0.1f;
        out[(((size_t)b * (ND_ * ND_) + dy * ND_ + dxi) * H_ + h) * W_ + w0 + m] = v;
    }
}

extern "C" void kernel_launch(void* const* d_in, const int* in_sizes, int n_in,
                              void* d_out, int out_size, void* d_ws, size_t ws_size,
                              hipStream_t stream) {
    const float* x1 = (const float*)d_in[0];
    const float* x2 = (const float*)d_in[1];
    float* out = (float*)d_out;
    dim3 grid(W_ / TW_, H_, B_ * NDG_);   // 8 x 96 x 24 = 18432 workgroups
    corr_wmma_kernel<<<grid, NT_, 0, stream>>>(x1, x2, out);
}
